// SuperGlue_17291538334076
// MI455X (gfx1250) — matrix operand order repaired and checked by reading, not compiled, not run on hardware
//
#include <hip/hip_runtime.h>
#include <hip/hip_bf16.h>

// ---------------------------------------------------------------------------
// SuperGlue forward for MI455X (gfx1250, wave32, WMMA).
// All GEMMs use v_wmma_f32_16x16x32_f16. fp32->fp16 conversion happens while
// staging tiles into LDS (packed ds_store_b128/b64). LDS tiles are
// double-buffered: next tile's global loads are issued before the WMMA block
// (acting as prefetch), and only ONE barrier per k-step is needed.
// Fragment reads are two 16B ds_load_b128 per lane matching the ISA A/B
// VGPR layouts (CDNA5 ISA 7.12.2).
// Sinkhorn couplings (8.4 MB) are L2-resident (192 MB L2).
// ---------------------------------------------------------------------------

typedef __attribute__((ext_vector_type(16))) _Float16 v16h;
typedef __attribute__((ext_vector_type(8)))  _Float16 v8h;
typedef __attribute__((ext_vector_type(4)))  _Float16 v4h;
typedef __attribute__((ext_vector_type(8)))  float    v8f;

union V16H { v16h v; v8h h[2]; };

__device__ inline v8f v8f_zero() {
  v8f z;
#pragma unroll
  for (int i = 0; i < 8; ++i) z[i] = 0.f;
  return z;
}

constexpr int Bb  = 2;     // batch
constexpr int Nn  = 1024;  // keypoints per image
constexpr int Dd  = 256;   // descriptor dim
constexpr int NHh = 4;     // heads
constexpr int DHd = 64;    // head dim
constexpr int Ll  = 18;    // GNN layers

// ===========================================================================
// GEMM: Y[b,o,n] = act( sum_i W[o,i] * X[b,i,n] + bias[o] )  (+BN/ReLU, +res)
// Block = 256 threads = 8 waves. Block tile 128(O) x 64(N); each wave owns a
// 16(O) row band and 4 WMMA n-subtiles. K-loop step 32, double-buffered LDS.
// ===========================================================================
template <bool BN_RELU, bool RES>
__global__ __launch_bounds__(256)
void gemm_wx(const float* __restrict__ W, const float* __restrict__ X,
             const float* __restrict__ bias, const float* __restrict__ bns,
             const float* __restrict__ bnb, const float* __restrict__ resid,
             float* __restrict__ Y, int Odim, int Idim, int Ndim) {
  constexpr int AS = 40;  // padded LDS stride (halves)
  __shared__ __align__(16) _Float16 lA[2][128 * AS];  // A row-major [o][k]
  __shared__ __align__(16) _Float16 lB[2][64 * AS];   // B col-major [n][k]
  const int t = threadIdx.x, lane = t & 31, wave = t >> 5;
  const int o0 = blockIdx.y * 128;
  const int n0 = blockIdx.x * 64;
  const int bz = blockIdx.z;
  const float* Xb = X + (size_t)bz * Idim * Ndim;
  float* Yb = Y + (size_t)bz * Odim * Ndim;
  const int l = lane & 15, hi = lane >> 4;

  // staging assignments
  const int rowA = t >> 1, cbA = (t & 1) * 16;   // A: 16 contiguous k
  const int ncB = t & 63, kcB = (t >> 6) * 8;    // B: 8-k chunk of one column

  float arA[16], arB[8];
  auto loadT = [&](int k0) {
    const float* sa = W + (size_t)(o0 + rowA) * Idim + (k0 + cbA);
#pragma unroll
    for (int j = 0; j < 16; ++j) arA[j] = sa[j];
    const float* sb = Xb + (size_t)(k0 + kcB) * Ndim + (n0 + ncB);
#pragma unroll
    for (int j = 0; j < 8; ++j) arB[j] = sb[(size_t)j * Ndim];
  };
  auto storeT = [&](int buf) {
    v8h r0, r1, rb;
#pragma unroll
    for (int j = 0; j < 8; ++j) {
      r0[j] = (_Float16)arA[j];
      r1[j] = (_Float16)arA[8 + j];
      rb[j] = (_Float16)arB[j];
    }
    v8h* dA = (v8h*)&lA[buf][rowA * AS + cbA];
    dA[0] = r0;
    dA[1] = r1;
    *(v8h*)&lB[buf][ncB * AS + kcB] = rb;
  };

  v8f acc[4];
#pragma unroll
  for (int s = 0; s < 4; ++s) acc[s] = v8f_zero();

  loadT(0);
  storeT(0);
  __syncthreads();

  int cur = 0;
  for (int k0 = 0; k0 < Idim; k0 += 32) {
    const bool more = (k0 + 32 < Idim);
    if (more) loadT(k0 + 32);  // overlaps with WMMA below

    V16H af;  // lanes 0-15: K{0..7,16..23}; lanes 16-31: K{8..15,24..31}
    const _Float16* pa = &lA[cur][(wave * 16 + l) * AS];
    af.h[0] = *(const v8h*)(pa + hi * 8);
    af.h[1] = *(const v8h*)(pa + 16 + hi * 8);
#pragma unroll
    for (int s = 0; s < 4; ++s) {
      V16H bf;  // lane=col: K{0..7,16..23} (lanes 0-15) / K{8..15,24..31} (lanes 16-31)
      const _Float16* pb = &lB[cur][(s * 16 + l) * AS + hi * 8];
      bf.h[0] = *(const v8h*)(pb);
      bf.h[1] = *(const v8h*)(pb + 16);
      acc[s] = __builtin_amdgcn_wmma_f32_16x16x32_f16(
          false, af.v, false, bf.v, (short)0, acc[s], false, false);
    }
    if (more) storeT(cur ^ 1);
    __syncthreads();  // single barrier per k-step (double-buffered)
    cur ^= 1;
  }

#pragma unroll
  for (int s = 0; s < 4; ++s) {
#pragma unroll
    for (int r = 0; r < 8; ++r) {
      const int orow = o0 + wave * 16 + (hi ? 8 + r : r);  // C/D layout
      const int ncol = n0 + s * 16 + l;
      float val = acc[s][r] + bias[orow];
      if (BN_RELU) val = fmaxf(val * bns[orow] + bnb[orow], 0.f);
      const size_t idx = (size_t)orow * Ndim + ncol;
      if (RES) val += resid[(size_t)bz * Odim * Ndim + idx];
      Yb[idx] = val;
    }
  }
}

// ===========================================================================
// S[z,n,m] = scale * sum_d Q[ch(d)][n] * K[ch(d)][m]
// Attention scores (cs=4N, Ktot=64, NHp=4) and the final cost matrix
// (cs=N, Ktot=256, NHp=1, outStride=1025 writing straight into couplings).
// Block tile 64(n) x 64(m), waves 4x2, 2 accumulators; double-buffered LDS.
// ===========================================================================
__global__ __launch_bounds__(256)
void gemm_qk(const float* __restrict__ Q, const float* __restrict__ Kk,
             float* __restrict__ Out, int Ndim, int cs, int Ktot, int NHp,
             int headStride, long long outZ, int outStride, float scale) {
  constexpr int AS = 40;
  __shared__ __align__(16) _Float16 lA[2][64 * AS];  // Q^T tile, col-major n
  __shared__ __align__(16) _Float16 lB[2][64 * AS];  // K tile, col-major m
  const int t = threadIdx.x, lane = t & 31, wave = t >> 5;
  const int wn = wave & 3, wm = wave >> 2;
  const int n0 = blockIdx.y * 64, m0 = blockIdx.x * 64;
  const int z = blockIdx.z, b = z / NHp, h = z - b * NHp;
  const float* qb = Q + (size_t)b * Dd * Ndim + (size_t)h * headStride;
  const float* kb = Kk + (size_t)b * Dd * Ndim + (size_t)h * headStride;
  float* out = Out + (size_t)z * (size_t)outZ;
  const int l = lane & 15, hi = lane >> 4;

  const int nc = t & 63, dc = (t >> 6) * 8;  // thread owns 8-d chunk of a col

  float arA[8], arB[8];
  auto loadT = [&](int d0) {
    const float* sa = qb + (size_t)(d0 + dc) * cs + (n0 + nc);
    const float* sb = kb + (size_t)(d0 + dc) * cs + (m0 + nc);
#pragma unroll
    for (int j = 0; j < 8; ++j) {
      arA[j] = sa[(size_t)j * cs];
      arB[j] = sb[(size_t)j * cs];
    }
  };
  auto storeT = [&](int buf) {
    v8h ra, rb;
#pragma unroll
    for (int j = 0; j < 8; ++j) {
      ra[j] = (_Float16)arA[j];
      rb[j] = (_Float16)arB[j];
    }
    *(v8h*)&lA[buf][nc * AS + dc] = ra;
    *(v8h*)&lB[buf][nc * AS + dc] = rb;
  };

  v8f acc[2];
  acc[0] = v8f_zero();
  acc[1] = v8f_zero();

  loadT(0);
  storeT(0);
  __syncthreads();

  int cur = 0;
  for (int d0 = 0; d0 < Ktot; d0 += 32) {
    const bool more = (d0 + 32 < Ktot);
    if (more) loadT(d0 + 32);

    V16H af;
    const _Float16* pa = &lA[cur][(wn * 16 + l) * AS];
    af.h[0] = *(const v8h*)(pa + hi * 8);
    af.h[1] = *(const v8h*)(pa + 16 + hi * 8);
#pragma unroll
    for (int s = 0; s < 2; ++s) {
      V16H bf;
      const _Float16* pb = &lB[cur][(wm * 32 + s * 16 + l) * AS + hi * 8];
      bf.h[0] = *(const v8h*)(pb);
      bf.h[1] = *(const v8h*)(pb + 16);
      acc[s] = __builtin_amdgcn_wmma_f32_16x16x32_f16(
          false, af.v, false, bf.v, (short)0, acc[s], false, false);
    }
    if (more) storeT(cur ^ 1);
    __syncthreads();
    cur ^= 1;
  }
#pragma unroll
  for (int s = 0; s < 2; ++s) {
#pragma unroll
    for (int r = 0; r < 8; ++r) {
      const int nr = n0 + wn * 16 + (hi ? 8 + r : r);
      const int mc = m0 + wm * 32 + s * 16 + l;
      out[(size_t)nr * outStride + mc] = acc[s][r] * scale;
    }
  }
}

// ===========================================================================
// msg[b, d*4+h, n] = sum_m P[z,n,m] * V[b, d*4+h, m]   (z = b*4+h)
// A = V rows (row-major in m), B = P^T (row n of P is column n of B).
// Block tile 64(d) x 32(n); waves 4x2; K = 1024; double-buffered LDS.
// ===========================================================================
__global__ __launch_bounds__(256)
void gemm_pv(const float* __restrict__ P, const float* __restrict__ V,
             float* __restrict__ MSG, int Ndim) {
  constexpr int AS = 40;
  __shared__ __align__(16) _Float16 lA[2][64 * AS];  // V tile row-major [d][mk]
  __shared__ __align__(16) _Float16 lB[2][32 * AS];  // P^T tile col-major [n][mk]
  const int t = threadIdx.x, lane = t & 31, wave = t >> 5;
  const int wd = wave & 3, wn = wave >> 2;
  const int n0 = blockIdx.x * 32;
  const int z = blockIdx.z, b = z >> 2, h = z & 3;
  const float* Pz = P + (size_t)z * Ndim * Ndim;
  const float* vb = V + (size_t)b * Dd * Ndim + (size_t)h * Ndim;
  float* msgb = MSG + (size_t)b * Dd * Ndim + (size_t)h * Ndim;
  const int l = lane & 15, hi = lane >> 4;

  const int dA = t >> 2, mbA = (t & 3) * 8;  // V: 8 contiguous m
  const int nB = t >> 3, mbB = (t & 7) * 4;  // P: 4 contiguous m

  float arA[8], arB[4];
  auto loadT = [&](int m0) {
    const float* sa = vb + (size_t)dA * 4 * Ndim + (m0 + mbA);
#pragma unroll
    for (int j = 0; j < 8; ++j) arA[j] = sa[j];
    const float* sb = Pz + (size_t)(n0 + nB) * Ndim + (m0 + mbB);
#pragma unroll
    for (int j = 0; j < 4; ++j) arB[j] = sb[j];
  };
  auto storeT = [&](int buf) {
    v8h ra;
#pragma unroll
    for (int j = 0; j < 8; ++j) ra[j] = (_Float16)arA[j];
    *(v8h*)&lA[buf][dA * AS + mbA] = ra;
    v4h rb;
#pragma unroll
    for (int j = 0; j < 4; ++j) rb[j] = (_Float16)arB[j];
    *(v4h*)&lB[buf][nB * AS + mbB] = rb;
  };

  v8f acc = v8f_zero();

  loadT(0);
  storeT(0);
  __syncthreads();

  int cur = 0;
  for (int m0 = 0; m0 < Ndim; m0 += 32) {
    const bool more = (m0 + 32 < Ndim);
    if (more) loadT(m0 + 32);

    V16H af;
    const _Float16* pa = &lA[cur][(wd * 16 + l) * AS];
    af.h[0] = *(const v8h*)(pa + hi * 8);
    af.h[1] = *(const v8h*)(pa + 16 + hi * 8);
    V16H bf;
    const _Float16* pb = &lB[cur][(wn * 16 + l) * AS + hi * 8];
    bf.h[0] = *(const v8h*)(pb);
    bf.h[1] = *(const v8h*)(pb + 16);
    acc = __builtin_amdgcn_wmma_f32_16x16x32_f16(
        false, af.v, false, bf.v, (short)0, acc, false, false);

    if (more) storeT(cur ^ 1);
    __syncthreads();
    cur ^= 1;
  }
#pragma unroll
  for (int r = 0; r < 8; ++r) {
    const int d = wd * 16 + (hi ? 8 + r : r);
    const int nc = n0 + wn * 16 + l;
    msgb[(size_t)d * 4 * Ndim + nc] = acc[r];
  }
}

// ===========================================================================
// Keypoint encoder: one block per point, activations staged in LDS.
// Writes dout[b,c,n] = desc[b,n,c] + enc[c] (fused descriptor transpose).
// ===========================================================================
__global__ __launch_bounds__(256)
void kenc_kernel(const float* __restrict__ kpts, const float* __restrict__ scr,
                 const float* __restrict__ sz, const float* __restrict__ desc,
                 const float* __restrict__ W0, const float* __restrict__ b0,
                 const float* __restrict__ s0, const float* __restrict__ bb0,
                 const float* __restrict__ W1, const float* __restrict__ b1,
                 const float* __restrict__ s1, const float* __restrict__ bb1,
                 const float* __restrict__ W2, const float* __restrict__ b2,
                 const float* __restrict__ s2, const float* __restrict__ bb2,
                 const float* __restrict__ Wo, const float* __restrict__ bo,
                 float* __restrict__ dout, int Ndim) {
  __shared__ float xin[3];
  __shared__ float bufA[128];
  __shared__ float bufB[128];
  const int n = blockIdx.x, b = blockIdx.y, t = threadIdx.x;
  if (t == 0) {
    const float sx = sz[b * 2 + 0], sy = sz[b * 2 + 1];
    const float sc = fmaxf(sx, sy) * 0.7f;
    xin[0] = (kpts[((size_t)b * Ndim + n) * 2 + 0] - sx * 0.5f) / sc;
    xin[1] = (kpts[((size_t)b * Ndim + n) * 2 + 1] - sy * 0.5f) / sc;
    xin[2] = scr[(size_t)b * Ndim + n];
  }
  __syncthreads();
  if (t < 32) {
    float a = b0[t];
    for (int i = 0; i < 3; ++i) a += W0[t * 3 + i] * xin[i];
    bufA[t] = fmaxf(a * s0[t] + bb0[t], 0.f);
  }
  __syncthreads();
  if (t < 64) {
    float a = b1[t];
    for (int i = 0; i < 32; ++i) a += W1[t * 32 + i] * bufA[i];
    bufB[t] = fmaxf(a * s1[t] + bb1[t], 0.f);
  }
  __syncthreads();
  if (t < 128) {
    float a = b2[t];
    for (int i = 0; i < 64; ++i) a += W2[t * 64 + i] * bufB[i];
    bufA[t] = fmaxf(a * s2[t] + bb2[t], 0.f);
  }
  __syncthreads();
  float a = bo[t];
  for (int i = 0; i < 128; ++i) a += Wo[t * 128 + i] * bufA[i];
  dout[((size_t)b * Dd + t) * Ndim + n] =
      desc[((size_t)b * Ndim + n) * Dd + t] + a;
}

// ========================= memory-bound helpers ============================

__global__ __launch_bounds__(256)
void softmax_rows(float* __restrict__ P, int Ndim) {
  float* row = P + (size_t)blockIdx.z * Ndim * Ndim + (size_t)blockIdx.x * Ndim;
  __shared__ float red[256];
  const int t = threadIdx.x;
  float m = -1e30f;
  for (int j = t; j < Ndim; j += 256) m = fmaxf(m, row[j]);
  red[t] = m;
  __syncthreads();
  for (int s = 128; s > 0; s >>= 1) {
    if (t < s) red[t] = fmaxf(red[t], red[t + s]);
    __syncthreads();
  }
  m = red[0];
  __syncthreads();
  float sum = 0.f;
  for (int j = t; j < Ndim; j += 256) {
    const float e = __expf(row[j] - m);
    row[j] = e;
    sum += e;
  }
  red[t] = sum;
  __syncthreads();
  for (int s = 128; s > 0; s >>= 1) {
    if (t < s) red[t] += red[t + s];
    __syncthreads();
  }
  const float inv = 1.0f / red[0];
  for (int j = t; j < Ndim; j += 256) row[j] *= inv;
}

__global__ __launch_bounds__(256)
void concat2(const float* __restrict__ X, const float* __restrict__ M,
             float* __restrict__ H, int Ndim) {
  const size_t i = (size_t)blockIdx.x * 256 + threadIdx.x;
  const size_t per = (size_t)2 * Dd * Ndim;
  const int b = (int)(i / per);
  const size_t rem = i - (size_t)b * per;
  const int c = (int)(rem / Ndim);
  const int n = (int)(rem - (size_t)c * Ndim);
  H[i] = (c < Dd) ? X[((size_t)b * Dd + c) * Ndim + n]
                  : M[((size_t)b * Dd + (c - Dd)) * Ndim + n];
}

__global__ __launch_bounds__(256)
void zero_f(float* __restrict__ p, int n) {
  const int i = blockIdx.x * 256 + threadIdx.x;
  if (i < n) p[i] = 0.f;
}

__global__ __launch_bounds__(256)
void fill_borders(float* __restrict__ C, const float* __restrict__ alpha, int MN) {
  const float al = alpha[0];
  const int idx = blockIdx.x * 256 + threadIdx.x;
  float* Cb = C + (size_t)blockIdx.y * (MN + 1) * (MN + 1);
  if (idx < MN) Cb[(size_t)idx * (MN + 1) + MN] = al;
  else if (idx < 2 * MN) Cb[(size_t)MN * (MN + 1) + (idx - MN)] = al;
  else if (idx == 2 * MN) Cb[(size_t)MN * (MN + 1) + MN] = al;
}

// u = log_mu - logsumexp_m(C + v[m]) : one block per row, two-pass LSE
__global__ __launch_bounds__(256)
void sink_u(const float* __restrict__ C, const float* __restrict__ Vv,
            float* __restrict__ U, int MN) {
  const int r = blockIdx.x, b = blockIdx.y, t = threadIdx.x;
  const float* row = C + ((size_t)b * (MN + 1) + r) * (MN + 1);
  const float* vv = Vv + (size_t)b * (MN + 1);
  __shared__ float red[256];
  float m = -1e30f;
  for (int j = t; j <= MN; j += 256) m = fmaxf(m, row[j] + vv[j]);
  red[t] = m;
  __syncthreads();
  for (int s = 128; s > 0; s >>= 1) {
    if (t < s) red[t] = fmaxf(red[t], red[t + s]);
    __syncthreads();
  }
  m = red[0];
  __syncthreads();
  float sum = 0.f;
  for (int j = t; j <= MN; j += 256) sum += __expf(row[j] + vv[j] - m);
  red[t] = sum;
  __syncthreads();
  for (int s = 128; s > 0; s >>= 1) {
    if (t < s) red[t] += red[t + s];
    __syncthreads();
  }
  if (t == 0) {
    const float norm = -logf((float)(2 * MN));
    const float logmu = (r < MN) ? norm : (logf((float)MN) + norm);
    U[(size_t)b * (MN + 1) + r] = logmu - (m + logf(red[0]));
  }
}

// v = log_nu - logsumexp_n(C + u[n]) : thread-per-column online LSE (coalesced)
__global__ __launch_bounds__(256)
void sink_v(const float* __restrict__ C, const float* __restrict__ U,
            float* __restrict__ Vv, int MN) {
  const int j = blockIdx.x * 256 + threadIdx.x;
  const int b = blockIdx.y;
  if (j > MN) return;
  const float* Cb = C + (size_t)b * (MN + 1) * (MN + 1);
  const float* uu = U + (size_t)b * (MN + 1);
  float m = -1e30f, s = 0.f;
  for (int i = 0; i <= MN; ++i) {
    const float a = Cb[(size_t)i * (MN + 1) + j] + uu[i];
    if (a > m) { s = s * __expf(m - a) + 1.f; m = a; }
    else s += __expf(a - m);
  }
  const float norm = -logf((float)(2 * MN));
  const float lognu = (j < MN) ? norm : (logf((float)MN) + norm);
  Vv[(size_t)b * (MN + 1) + j] = lognu - (m + logf(s));
}

__global__ __launch_bounds__(256)
void write_Z(const float* __restrict__ C, const float* __restrict__ U,
             const float* __restrict__ Vv, float* __restrict__ Z, int MN) {
  const size_t total = (size_t)(MN + 1) * (MN + 1);
  const size_t i = (size_t)blockIdx.x * 256 + threadIdx.x;
  if (i >= total) return;
  const int b = blockIdx.y;
  const int r = (int)(i / (MN + 1));
  const int c = (int)(i % (MN + 1));
  const float norm = -logf((float)(2 * MN));
  Z[(size_t)b * total + i] = C[(size_t)b * total + i] +
                             U[(size_t)b * (MN + 1) + r] +
                             Vv[(size_t)b * (MN + 1) + c] - norm;
}

__global__ __launch_bounds__(256)
void row_argmax(const float* __restrict__ Z, float* __restrict__ max0,
                int* __restrict__ m0r, int MN) {
  const int r = blockIdx.x, b = blockIdx.y, t = threadIdx.x;
  const float* row = Z + ((size_t)b * (MN + 1) + r) * (MN + 1);
  __shared__ float rm[256];
  __shared__ int ri[256];
  float m = -1e30f;
  int mi = 0;
  for (int j = t; j < MN; j += 256) {
    const float v = row[j];
    if (v > m) { m = v; mi = j; }
  }
  rm[t] = m; ri[t] = mi;
  __syncthreads();
  for (int s = 128; s > 0; s >>= 1) {
    if (t < s) {
      if (rm[t + s] > rm[t] || (rm[t + s] == rm[t] && ri[t + s] < ri[t])) {
        rm[t] = rm[t + s]; ri[t] = ri[t + s];
      }
    }
    __syncthreads();
  }
  if (t == 0) {
    max0[(size_t)b * MN + r] = rm[0];
    m0r[(size_t)b * MN + r] = ri[0];
  }
}

__global__ __launch_bounds__(256)
void col_argmax(const float* __restrict__ Z, int* __restrict__ m1r, int MN) {
  const int j = blockIdx.x * 256 + threadIdx.x;
  const int b = blockIdx.y;
  if (j >= MN) return;
  const float* Zb = Z + (size_t)b * (MN + 1) * (MN + 1);
  float m = -1e30f;
  int mi = 0;
  for (int i = 0; i < MN; ++i) {
    const float v = Zb[(size_t)i * (MN + 1) + j];
    if (v > m) { m = v; mi = i; }
  }
  m1r[(size_t)b * MN + j] = mi;
}

__global__ __launch_bounds__(256)
void match0(const float* __restrict__ max0, const int* __restrict__ m0r,
            const int* __restrict__ m1r, int* __restrict__ m0o,
            float* __restrict__ ms0, int* __restrict__ valid0, int MN) {
  const int i = blockIdx.x * 256 + threadIdx.x;
  const int b = blockIdx.y;
  if (i >= MN) return;
  const int mo = m0r[(size_t)b * MN + i];
  const bool mutual = (m1r[(size_t)b * MN + mo] == i);
  const float ms = mutual ? __expf(max0[(size_t)b * MN + i]) : 0.f;
  const bool v = mutual && (ms > 0.2f);
  ms0[(size_t)b * MN + i] = ms;
  valid0[(size_t)b * MN + i] = v ? 1 : 0;
  m0o[(size_t)b * MN + i] = v ? mo : -1;
}

__global__ __launch_bounds__(256)
void match1(const int* __restrict__ m0r, const int* __restrict__ m1r,
            const float* __restrict__ ms0, const int* __restrict__ valid0,
            int* __restrict__ m1o, float* __restrict__ ms1, int MN) {
  const int j = blockIdx.x * 256 + threadIdx.x;
  const int b = blockIdx.y;
  if (j >= MN) return;
  const int mi = m1r[(size_t)b * MN + j];
  const bool mutual = (m0r[(size_t)b * MN + mi] == j);
  const float ms = mutual ? ms0[(size_t)b * MN + mi] : 0.f;
  const bool v = mutual && (valid0[(size_t)b * MN + mi] != 0);
  ms1[(size_t)b * MN + j] = ms;
  m1o[(size_t)b * MN + j] = v ? mi : -1;
}

// ============================== host side ==================================

static void propagate(const float* x, const float* src, const float* qw,
                      const float* qb_, const float* kw, const float* kb_,
                      const float* vw, const float* vb_, const float* mw,
                      const float* mb_, const float* w1, const float* b1_,
                      const float* bns, const float* bnb, const float* w2,
                      const float* b2_, float* Qb, float* Kb, float* Vb,
                      float* PROB, float* MSG, float* MSG2, float* H,
                      float* H2, float* out, hipStream_t stream) {
  const dim3 blk(256);
  gemm_wx<false, false><<<dim3(Nn / 64, Dd / 128, Bb), blk, 0, stream>>>(
      qw, x, qb_, nullptr, nullptr, nullptr, Qb, Dd, Dd, Nn);
  gemm_wx<false, false><<<dim3(Nn / 64, Dd / 128, Bb), blk, 0, stream>>>(
      kw, src, kb_, nullptr, nullptr, nullptr, Kb, Dd, Dd, Nn);
  gemm_wx<false, false><<<dim3(Nn / 64, Dd / 128, Bb), blk, 0, stream>>>(
      vw, src, vb_, nullptr, nullptr, nullptr, Vb, Dd, Dd, Nn);
  gemm_qk<<<dim3(Nn / 64, Nn / 64, Bb * NHh), blk, 0, stream>>>(
      Qb, Kb, PROB, Nn, 4 * Nn, DHd, NHh, Nn, (long long)Nn * Nn, Nn, 0.125f);
  softmax_rows<<<dim3(Nn, 1, Bb * NHh), blk, 0, stream>>>(PROB, Nn);
  gemm_pv<<<dim3(Nn / 32, 1, Bb * NHh), blk, 0, stream>>>(PROB, Vb, MSG, Nn);
  gemm_wx<false, false><<<dim3(Nn / 64, Dd / 128, Bb), blk, 0, stream>>>(
      mw, MSG, mb_, nullptr, nullptr, nullptr, MSG2, Dd, Dd, Nn);
  concat2<<<dim3((unsigned)((size_t)Bb * 2 * Dd * Nn / 256)), blk, 0, stream>>>(
      x, MSG2, H, Nn);
  gemm_wx<true, false><<<dim3(Nn / 64, (2 * Dd) / 128, Bb), blk, 0, stream>>>(
      w1, H, b1_, bns, bnb, nullptr, H2, 2 * Dd, 2 * Dd, Nn);
  gemm_wx<false, true><<<dim3(Nn / 64, Dd / 128, Bb), blk, 0, stream>>>(
      w2, H2, b2_, nullptr, nullptr, x, out, Dd, 2 * Dd, Nn);
}

extern "C" void kernel_launch(void* const* d_in, const int* in_sizes, int n_in,
                              void* d_out, int out_size, void* d_ws,
                              size_t ws_size, hipStream_t stream) {
  (void)in_sizes; (void)n_in; (void)out_size; (void)ws_size;
  // Inputs, then params flattened as a JAX pytree (dicts sorted by key):
  // params: bin_score, fb, fw, gnn{b1,b2,bn_b,bn_s,kb,kw,mb,mw,qb,qw,vb,vw,w1,w2},
  //         kenc{hidden[(W,b,s,bb)x3], out(W,b)}
  const float* kpts0 = (const float*)d_in[0];
  const float* kpts1 = (const float*)d_in[1];
  const float* ksc0  = (const float*)d_in[2];
  const float* ksc1  = (const float*)d_in[3];
  const float* desc0 = (const float*)d_in[4];
  const float* desc1 = (const float*)d_in[5];
  const float* size0 = (const float*)d_in[6];
  const float* size1 = (const float*)d_in[7];
  const float* bin_score = (const float*)d_in[8];
  const float* fb    = (const float*)d_in[9];
  const float* fw    = (const float*)d_in[10];
  const float* g_b1  = (const float*)d_in[11];
  const float* g_b2  = (const float*)d_in[12];
  const float* g_bnb = (const float*)d_in[13];
  const float* g_bns = (const float*)d_in[14];
  const float* g_kb  = (const float*)d_in[15];
  const float* g_kw  = (const float*)d_in[16];
  const float* g_mb  = (const float*)d_in[17];
  const float* g_mw  = (const float*)d_in[18];
  const float* g_qb  = (const float*)d_in[19];
  const float* g_qw  = (const float*)d_in[20];
  const float* g_vb  = (const float*)d_in[21];
  const float* g_vw  = (const float*)d_in[22];
  const float* g_w1  = (const float*)d_in[23];
  const float* g_w2  = (const float*)d_in[24];
  const float* kW0 = (const float*)d_in[25];
  const float* kb0 = (const float*)d_in[26];
  const float* ks0 = (const float*)d_in[27];
  const float* kbb0 = (const float*)d_in[28];
  const float* kW1 = (const float*)d_in[29];
  const float* kb1 = (const float*)d_in[30];
  const float* ks1 = (const float*)d_in[31];
  const float* kbb1 = (const float*)d_in[32];
  const float* kW2 = (const float*)d_in[33];
  const float* kb2 = (const float*)d_in[34];
  const float* ks2 = (const float*)d_in[35];
  const float* kbb2 = (const float*)d_in[36];
  const float* kWo = (const float*)d_in[37];
  const float* kbo = (const float*)d_in[38];

  // Workspace layout (floats)
  const size_t SD = (size_t)Bb * Dd * Nn;         // 512K
  const size_t SH = (size_t)Bb * 2 * Dd * Nn;     // 1M
  const size_t SP = (size_t)Bb * NHh * Nn * Nn;   // 8M
  const size_t SC = (size_t)Bb * 1025 * 1025;     // ~2.1M
  float* ws   = (float*)d_ws;
  float* D0A  = ws;            ws += SD;
  float* D0B  = ws;            ws += SD;
  float* D1A  = ws;            ws += SD;
  float* D1B  = ws;            ws += SD;
  float* Qb   = ws;            ws += SD;
  float* Kb   = ws;            ws += SD;
  float* Vb   = ws;            ws += SD;
  float* MSG  = ws;            ws += SD;
  float* MSG2 = ws;            ws += SD;
  float* H    = ws;            ws += SH;
  float* H2   = ws;            ws += SH;
  float* PROB = ws;            ws += SP;
  float* COUP = ws;            ws += SC;
  float* U    = ws;            ws += (size_t)Bb * 1025;
  float* Vv   = ws;            ws += (size_t)Bb * 1025;
  float* MAX0 = ws;            ws += (size_t)Bb * Nn;
  int* M0R  = (int*)ws;        ws += (size_t)Bb * Nn;
  int* M1R  = (int*)ws;        ws += (size_t)Bb * Nn;
  int* VAL0 = (int*)ws;        ws += (size_t)Bb * Nn;

  const dim3 blk(256);

  // Keypoint encoder (+ descriptor transpose + add)
  kenc_kernel<<<dim3(Nn, Bb), blk, 0, stream>>>(
      kpts0, ksc0, size0, desc0, kW0, kb0, ks0, kbb0, kW1, kb1, ks1, kbb1,
      kW2, kb2, ks2, kbb2, kWo, kbo, D0A, Nn);
  kenc_kernel<<<dim3(Nn, Bb), blk, 0, stream>>>(
      kpts1, ksc1, size1, desc1, kW0, kb0, ks0, kbb0, kW1, kb1, ks1, kbb1,
      kW2, kb2, ks2, kbb2, kWo, kbo, D1A, Nn);

  float* cur0 = D0A; float* cur1 = D1A;
  float* nxt0 = D0B; float* nxt1 = D1B;
  for (int i = 0; i < Ll; ++i) {
    const size_t wOff = (size_t)i * Dd * Dd;
    const size_t w1Off = (size_t)i * (2 * Dd) * (2 * Dd);
    const size_t w2Off = (size_t)i * Dd * (2 * Dd);
    const float* qw = g_qw + wOff;  const float* qb_ = g_qb + (size_t)i * Dd;
    const float* kw = g_kw + wOff;  const float* kb_ = g_kb + (size_t)i * Dd;
    const float* vw = g_vw + wOff;  const float* vb_ = g_vb + (size_t)i * Dd;
    const float* mw = g_mw + wOff;  const float* mb_ = g_mb + (size_t)i * Dd;
    const float* w1 = g_w1 + w1Off; const float* b1_ = g_b1 + (size_t)i * 2 * Dd;
    const float* bns = g_bns + (size_t)i * 2 * Dd;
    const float* bnb = g_bnb + (size_t)i * 2 * Dd;
    const float* w2 = g_w2 + w2Off; const float* b2_ = g_b2 + (size_t)i * Dd;
    const float* src0 = (i % 2 == 0) ? cur0 : cur1;  // self vs cross
    const float* src1 = (i % 2 == 0) ? cur1 : cur0;
    propagate(cur0, src0, qw, qb_, kw, kb_, vw, vb_, mw, mb_, w1, b1_, bns,
              bnb, w2, b2_, Qb, Kb, Vb, PROB, MSG, MSG2, H, H2, nxt0, stream);
    propagate(cur1, src1, qw, qb_, kw, kb_, vw, vb_, mw, mb_, w1, b1_, bns,
              bnb, w2, b2_, Qb, Kb, Vb, PROB, MSG, MSG2, H, H2, nxt1, stream);
    float* t0 = cur0; cur0 = nxt0; nxt0 = t0;
    float* t1 = cur1; cur1 = nxt1; nxt1 = t1;
  }

  // Final projection + cost matrix straight into couplings' inner block
  gemm_wx<false, false><<<dim3(Nn / 64, Dd / 128, Bb), blk, 0, stream>>>(
      fw, cur0, fb, nullptr, nullptr, nullptr, Qb, Dd, Dd, Nn);
  gemm_wx<false, false><<<dim3(Nn / 64, Dd / 128, Bb), blk, 0, stream>>>(
      fw, cur1, fb, nullptr, nullptr, nullptr, Kb, Dd, Dd, Nn);
  gemm_qk<<<dim3(Nn / 64, Nn / 64, Bb), blk, 0, stream>>>(
      Qb, Kb, COUP, Nn, Nn, Dd, 1, 0, (long long)1025 * 1025, 1025, 0.0625f);
  fill_borders<<<dim3(9, Bb), blk, 0, stream>>>(COUP, bin_score, Nn);

  // Sinkhorn (couplings fit in L2: 8.4 MB << 192 MB)
  zero_f<<<dim3((2 * Bb * 1025 + 255) / 256), blk, 0, stream>>>(U, 2 * Bb * 1025);
  for (int it = 0; it < 50; ++it) {
    sink_u<<<dim3(Nn + 1, Bb), blk, 0, stream>>>(COUP, Vv, U, Nn);
    sink_v<<<dim3((Nn + 1 + 255) / 256, Bb), blk, 0, stream>>>(COUP, U, Vv, Nn);
  }

  // Outputs: Z | m0(int) | m1(int) | mscores0 | mscores1
  float* Zout = (float*)d_out;
  const size_t ZTOT = (size_t)1025 * 1025;
  int* m0o = (int*)(Zout + (size_t)Bb * ZTOT);
  int* m1o = m0o + (size_t)Bb * Nn;
  float* ms0 = (float*)(m1o + (size_t)Bb * Nn);
  float* ms1 = ms0 + (size_t)Bb * Nn;

  write_Z<<<dim3((unsigned)((ZTOT + 255) / 256), Bb), blk, 0, stream>>>(
      COUP, U, Vv, Zout, Nn);
  row_argmax<<<dim3(Nn, Bb), blk, 0, stream>>>(Zout, MAX0, M0R, Nn);
  col_argmax<<<dim3(Nn / 256, Bb), blk, 0, stream>>>(Zout, M1R, Nn);
  match0<<<dim3(Nn / 256, Bb), blk, 0, stream>>>(MAX0, M0R, M1R, m0o, ms0,
                                                 VAL0, Nn);
  match1<<<dim3(Nn / 256, Bb), blk, 0, stream>>>(M0R, M1R, ms0, VAL0, m1o,
                                                 ms1, Nn);
}
